// SelfAttention_25563645346656
// MI455X (gfx1250) — matrix operand order repaired and checked by reading, not compiled, not run on hardware
//
#include <hip/hip_runtime.h>
#include <hip/hip_bf16.h>

typedef __attribute__((ext_vector_type(16))) __bf16 v16bf;
typedef __attribute__((ext_vector_type(8)))  __bf16 v8bf;
typedef __attribute__((ext_vector_type(8)))  float  v8f;
typedef __attribute__((ext_vector_type(4)))  float  v4f;

#define WMMA_BF16(A, B, C) \
    __builtin_amdgcn_wmma_f32_16x16x32_bf16(false, (A), false, (B), (short)0, (C), false, false)

#if defined(__AMDGCN__) && __has_builtin(__builtin_amdgcn_tensor_load_to_lds)
#define HAVE_TDM 1
#else
#define HAVE_TDM 0
#endif

constexpr int B_ = 4, S_ = 4096, D_ = 768, U_ = 64;
constexpr int M_ = B_ * S_;        // 16384 flattened (b,s) rows
constexpr int BQ = 128;            // queries per workgroup (16 per wave x 8 waves)
constexpr int BK = 64;             // keys per block
constexpr int BD = 192;            // d-columns per workgroup (12 x 16 tiles)
constexpr int PAD_STRIDE = 72;     // 64 elems + 8 pad (TDM: 4-DWORD pad per 32-DWORD row)
constexpr int VT_ELEMS = BD * PAD_STRIDE;  // 13824 bf16 per V buffer
constexpr int KT_ELEMS = BK * PAD_STRIDE;  // 4608 bf16 per K buffer

#if HAVE_TDM
typedef __attribute__((ext_vector_type(4))) unsigned int u32x4;
typedef __attribute__((ext_vector_type(8))) int i32x8;
typedef __attribute__((ext_vector_type(4))) int i32x4;

// 2-D bf16 tile -> LDS via Tensor Data Mover.  Row = tile_x elems (must be
// 128B here: one pad interval of 32 DWORDs), hardware appends 4 DWORDs of pad
// per row -> LDS row stride 72 elements (bank-conflict-free b128 reads).
__device__ __forceinline__ void tdm_load(unsigned lds_addr, const void* gsrc,
                                         unsigned tile_x, unsigned tile_y,
                                         unsigned stride_elems,
                                         unsigned tensor_x, unsigned tensor_y) {
    unsigned long long ga = (unsigned long long)(uintptr_t)gsrc;
    u32x4 g0;
    g0[0] = 1u;                                            // count=1 (valid D#)
    g0[1] = lds_addr;                                      // LDS byte address
    g0[2] = (unsigned)ga;                                  // global_addr[31:0]
    g0[3] = (unsigned)((ga >> 32) & 0x01ffffffu) | (2u << 30); // addr[56:32] | type=2
    i32x8 g1;
    g1[0] = (1 << 16) | (1 << 20) | (4 << 22) | (3 << 25); // 2B data | pad_en | ivl=32DW | amt=4DW
    g1[1] = (int)((tensor_x & 0xffffu) << 16);             // bar_addr=0 | tensor_dim0 lo16
    g1[2] = (int)((tensor_x >> 16) | ((tensor_y & 0xffffu) << 16)); // dim0 hi | dim1 lo
    g1[3] = (int)((tensor_y >> 16) | (tile_x << 16));      // dim1 hi | tile_dim0
    g1[4] = (int)tile_y;                                   // tile_dim1 | tile_dim2=0
    g1[5] = (int)stride_elems;                             // tensor_dim0_stride lo32
    g1[6] = 0;
    g1[7] = 0;
    i32x4 z4 = {};
#if __clang_major__ >= 23
    i32x8 z8 = {};
    __builtin_amdgcn_tensor_load_to_lds(g0, g1, z4, z4, z8, 0);
#else
    __builtin_amdgcn_tensor_load_to_lds(g0, g1, z4, z4, 0);
#endif
}
#endif // HAVE_TDM

// ---------------------------------------------------------------------------
// Kernel 0: Wq/Wk -> bf16 [U][D] (WMMA B-operand contiguous loads).
// ---------------------------------------------------------------------------
__global__ void prep_w(const float* __restrict__ Wq, const float* __restrict__ Wk,
                       __bf16* __restrict__ wtq, __bf16* __restrict__ wtk) {
    int i = blockIdx.x * blockDim.x + threadIdx.x;
    if (i < U_ * D_) {
        int u = i / D_, k = i % D_;
        wtq[i] = (__bf16)Wq[k * U_ + u];
        wtk[i] = (__bf16)Wk[k * U_ + u];
    }
}

// ---------------------------------------------------------------------------
// Kernel 0b: x [b][s][d] fp32 -> xt [b][d][s] bf16 (tiled LDS transpose) so
// V^T tile rows are contiguous and TDM-copyable.
// ---------------------------------------------------------------------------
__global__ __launch_bounds__(256) void transpose_x(const float* __restrict__ x,
                                                   __bf16* __restrict__ xt) {
    __shared__ float t[32][33];
    const int tx = threadIdx.x & 31;
    const int ty = threadIdx.x >> 5;                 // 0..7
    const int s0 = blockIdx.x * 32;
    const int d0 = blockIdx.y * 32;
    const int b  = blockIdx.z;
#pragma unroll
    for (int ii = 0; ii < 4; ++ii) {
        int s = s0 + ty + ii * 8;
        t[ty + ii * 8][tx] = x[((size_t)b * S_ + s) * D_ + d0 + tx];
    }
    __syncthreads();
#pragma unroll
    for (int ii = 0; ii < 4; ++ii) {
        int d = d0 + ty + ii * 8;
        xt[((size_t)b * D_ + d) * S_ + s0 + tx] = (__bf16)t[tx][ty + ii * 8];
    }
}

// ---------------------------------------------------------------------------
// Kernel 1: q = x @ Wq, k = x @ Wk (bf16 WMMA, fp32 accumulate) -> bf16.
// ---------------------------------------------------------------------------
__global__ __launch_bounds__(256) void proj_qk(const float* __restrict__ x,
                                               const __bf16* __restrict__ wtq,
                                               const __bf16* __restrict__ wtk,
                                               __bf16* __restrict__ qb,
                                               __bf16* __restrict__ kb) {
    const int tid  = threadIdx.x;
    const int lane = tid & 31;
    const int wave = tid >> 5;
    const int lq   = lane & 15;
    const int half = lane >> 4;
    const int m0   = blockIdx.x * 128 + wave * 16;

    v8f accq[4] = {};
    v8f acck[4] = {};

    const int arow = m0 + lq;
    for (int k0 = 0; k0 < D_; k0 += 32) {
        v16bf ax;
        {
            const float* ap = x + (size_t)arow * D_ + k0 + half * 16;
            v4f a0 = *(const v4f*)(ap);
            v4f a1 = *(const v4f*)(ap + 4);
            v4f a2 = *(const v4f*)(ap + 8);
            v4f a3 = *(const v4f*)(ap + 12);
#pragma unroll
            for (int j = 0; j < 4; ++j) {
                ax[j]      = (__bf16)a0[j];
                ax[4 + j]  = (__bf16)a1[j];
                ax[8 + j]  = (__bf16)a2[j];
                ax[12 + j] = (__bf16)a3[j];
            }
        }
#pragma unroll
        for (int c = 0; c < 4; ++c) {
            const int u = c * 16 + lq;
            v16bf bwq = *(const v16bf*)(wtq + (size_t)u * D_ + k0 + half * 16);
            v16bf bwk = *(const v16bf*)(wtk + (size_t)u * D_ + k0 + half * 16);
            accq[c] = WMMA_BF16(ax, bwq, accq[c]);
            acck[c] = WMMA_BF16(ax, bwk, acck[c]);
        }
    }
#pragma unroll
    for (int c = 0; c < 4; ++c) {
        const int u = c * 16 + lq;
#pragma unroll
        for (int v = 0; v < 8; ++v) {
            const size_t row = (size_t)(m0 + half * 8 + v);
            qb[row * U_ + u] = (__bf16)accq[c][v];
            kb[row * U_ + u] = (__bf16)acck[c][v];
        }
    }
}

// ---------------------------------------------------------------------------
// Kernel 2: flash attention.  Grid (D/BD, S/BQ, B); 8 waves x 16 queries.
// S^T = K @ Q^T so P lands in the A-operand layout of O = P @ V.  K and V^T
// blocks are staged into padded LDS tiles by the TDM (double-buffered, issued
// by wave 0, TENSORcnt-tracked) and consumed as WMMA fragments via ds_load.
// ---------------------------------------------------------------------------
__global__ __launch_bounds__(256) void flash_attn(const __bf16* __restrict__ qb,
                                                  const __bf16* __restrict__ kb,
                                                  const __bf16* __restrict__ xt,
                                                  const int* __restrict__ mask,
                                                  float* __restrict__ out) {
    __shared__ __bf16 Vsh[2 * VT_ELEMS]; // V^T tiles [d][72], 2 x 27648 B
    __shared__ __bf16 Ksh[2 * KT_ELEMS]; // K  tiles  [key][72], 2 x 9216 B

    const int tid  = threadIdx.x;
    const int lane = tid & 31;
    const int wave = tid >> 5;
    const int lq   = lane & 15;
    const int half = lane >> 4;

    const int dcol0 = blockIdx.x * BD;
    const int qrow0 = blockIdx.y * BQ;
    const int b     = blockIdx.z;
    const size_t rowbase = (size_t)b * S_;

    const __bf16* xt_t0 = xt + ((size_t)b * D_ + dcol0) * S_; // +kb0 per block
    const __bf16* kb_t0 = kb + rowbase * U_;                  // +kb0*U per block

    // Q fragments: B operand of S^T (lane = q column, contraction u contiguous)
    const size_t qrow = rowbase + qrow0 + wave * 16 + lq;
    v16bf bq[2];
#pragma unroll
    for (int ks = 0; ks < 2; ++ks)
        bq[ks] = *(const v16bf*)(qb + qrow * U_ + ks * 32 + half * 16);

    float m_run = -__builtin_inff();
    float l_run = 0.0f;
    v8f acc[12] = {};

    const int nblk = S_ / BK; // 64

#if HAVE_TDM
    const unsigned v_lds0 = (unsigned)(uintptr_t)(void*)&Vsh[0];
    const unsigned k_lds0 = (unsigned)(uintptr_t)(void*)&Ksh[0];
    if (wave == 0) {
        tdm_load(v_lds0, xt_t0, BK, BD, S_, S_, B_ * D_);
        tdm_load(k_lds0, kb_t0, U_, BK, U_, U_, M_);
    }
#endif

    for (int i = 0; i < nblk; ++i) {
        const int kb0 = i * BK;
#if HAVE_TDM
        if (wave == 0) {
            if (i + 1 < nblk) { // issue next block's DMA, then wait for current
                tdm_load(v_lds0 + ((i + 1) & 1) * (VT_ELEMS * 2),
                         xt_t0 + (i + 1) * BK, BK, BD, S_, S_, B_ * D_);
                tdm_load(k_lds0 + ((i + 1) & 1) * (KT_ELEMS * 2),
                         kb_t0 + (size_t)(i + 1) * BK * U_, U_, BK, U_, U_, M_);
                __builtin_amdgcn_s_wait_tensorcnt((short)2);
            } else {
                __builtin_amdgcn_s_wait_tensorcnt((short)0);
            }
        }
        __syncthreads(); // current buffers visible to all waves
#else
        { // manual staging fallback (same padded layouts)
            const __bf16* srcv = xt_t0 + kb0;
            __bf16* dstv = &Vsh[(i & 1) * VT_ELEMS];
            for (int j = tid; j < BD * BK; j += 256) {
                int ky = j & 63, d = j >> 6;
                dstv[d * PAD_STRIDE + ky] = srcv[(size_t)d * S_ + ky];
            }
            const __bf16* srck = kb_t0 + (size_t)kb0 * U_;
            __bf16* dstk = &Ksh[(i & 1) * KT_ELEMS];
            for (int j = tid; j < BK * U_; j += 256) {
                int u = j & 63, ky = j >> 6;
                dstk[ky * PAD_STRIDE + u] = srck[(size_t)ky * U_ + u];
            }
        }
        __syncthreads();
#endif
        const __bf16* Vbuf = &Vsh[(i & 1) * VT_ELEMS];
        const __bf16* Kbuf = &Ksh[(i & 1) * KT_ELEMS];

        // ---- S^T = K @ Q^T : 4 key tiles, contraction over U=64 ----
        float p[4][8];
        float mb = -3.0e38f;
#pragma unroll
        for (int t = 0; t < 4; ++t) {
            v8f s = {};
            const __bf16* kp = Kbuf + (t * 16 + lq) * PAD_STRIDE;
#pragma unroll
            for (int ks = 0; ks < 2; ++ks) {
                const __bf16* kpp = kp + ks * 32 + half * 16;
                v8bf k0 = *(const v8bf*)(kpp);
                v8bf k1 = *(const v8bf*)(kpp + 8);
                v16bf ak = __builtin_shufflevector(k0, k1, 0, 1, 2, 3, 4, 5, 6, 7,
                                                   8, 9, 10, 11, 12, 13, 14, 15);
                s = WMMA_BF16(ak, bq[ks], s);
            }
            const int* mp = mask + (size_t)b * S_ + kb0 + t * 16 + half * 8;
#pragma unroll
            for (int v = 0; v < 8; ++v) {
                float sv = (mp[v] != 0) ? s[v] * 0.125f : -1.0e9f;
                p[t][v]  = sv;
                mb       = fmaxf(mb, sv);
            }
        }

        // ---- online softmax (per q = lq; partner keys in lane^16) ----
        mb = fmaxf(mb, __shfl_xor(mb, 16, 32));
        const float m_new = fmaxf(m_run, mb);
        const float alpha = __expf(m_run - m_new);
        float rs = 0.0f;
#pragma unroll
        for (int t = 0; t < 4; ++t)
#pragma unroll
            for (int v = 0; v < 8; ++v) {
                float e = __expf(p[t][v] - m_new);
                p[t][v] = e;
                rs += e;
            }
        rs += __shfl_xor(rs, 16, 32);
        l_run = l_run * alpha + rs;
        m_run = m_new;

        float alphaV[8];
#pragma unroll
        for (int v = 0; v < 8; ++v)
            alphaV[v] = __shfl(alpha, half * 8 + v, 32);
#pragma unroll
        for (int dt = 0; dt < 12; ++dt)
#pragma unroll
            for (int v = 0; v < 8; ++v)
                acc[dt][v] *= alphaV[v];

        // pack P as A fragments (layout already matches)
        v16bf pa[2];
#pragma unroll
        for (int ks = 0; ks < 2; ++ks)
#pragma unroll
            for (int j = 0; j < 8; ++j) {
                pa[ks][j]     = (__bf16)p[2 * ks][j];
                pa[ks][8 + j] = (__bf16)p[2 * ks + 1][j];
            }

        // ---- O += P @ V from padded LDS V^T tile ----
#pragma unroll
        for (int dt = 0; dt < 12; ++dt) {
            const __bf16* vp = Vbuf + (dt * 16 + lq) * PAD_STRIDE;
#pragma unroll
            for (int ks = 0; ks < 2; ++ks) {
                const __bf16* q0 = vp + ks * 32 + half * 8;
                v8bf b0 = *(const v8bf*)(q0);
                v8bf b1 = *(const v8bf*)(q0 + 16);
                v16bf bv = __builtin_shufflevector(b0, b1, 0, 1, 2, 3, 4, 5, 6, 7,
                                                   8, 9, 10, 11, 12, 13, 14, 15);
                acc[dt] = WMMA_BF16(pa[ks], bv, acc[dt]);
            }
        }
        __syncthreads(); // all reads done before DMA overwrites this buffer
    }

    // ---- epilogue: 1/l, query mask, store ----
    const int q_me = qrow0 + wave * 16 + lq;
    const float fac = (mask[(size_t)b * S_ + q_me] != 0 ? 1.0f : 0.0f) / l_run;
    float facV[8];
#pragma unroll
    for (int v = 0; v < 8; ++v)
        facV[v] = __shfl(fac, half * 8 + v, 32);

#pragma unroll
    for (int dt = 0; dt < 12; ++dt) {
        const int d = dcol0 + dt * 16 + lq;
#pragma unroll
        for (int v = 0; v < 8; ++v) {
            const size_t q = (size_t)(qrow0 + wave * 16 + half * 8 + v);
            out[(rowbase + q) * D_ + d] = acc[dt][v] * facV[v];
        }
    }
}

// ---------------------------------------------------------------------------
extern "C" void kernel_launch(void* const* d_in, const int* in_sizes, int n_in,
                              void* d_out, int out_size, void* d_ws, size_t ws_size,
                              hipStream_t stream) {
    (void)in_sizes; (void)n_in; (void)out_size; (void)ws_size;
    const float* x    = (const float*)d_in[0];
    const int*   mask = (const int*)d_in[1];
    const float* Wq   = (const float*)d_in[2];
    const float* Wk   = (const float*)d_in[3];
    float* out = (float*)d_out;

    char* ws = (char*)d_ws;
    __bf16* wtq = (__bf16*)(ws);                    //  96 KB
    __bf16* wtk = (__bf16*)(ws + 98304);            //  96 KB
    __bf16* qb  = (__bf16*)(ws + 196608);           //   2 MB
    __bf16* kb  = (__bf16*)(ws + 2293760);          //   2 MB
    __bf16* xt  = (__bf16*)(ws + 4390912);          //  24 MB  [B][D][S] bf16

    prep_w<<<(U_ * D_ + 255) / 256, 256, 0, stream>>>(Wq, Wk, wtq, wtk);
    dim3 tg(S_ / 32, D_ / 32, B_);
    transpose_x<<<tg, 256, 0, stream>>>(x, xt);
    proj_qk<<<M_ / 128, 256, 0, stream>>>(x, wtq, wtk, qb, kb);
    dim3 grid(D_ / BD, S_ / BQ, B_);
    flash_attn<<<grid, 256, 0, stream>>>(qb, kb, xt, mask, out);
}
